// LTFGW_61787399520835
// MI455X (gfx1250) — matrix operand order repaired and checked by reading, not compiled, not run on hardware
//
#include <hip/hip_runtime.h>

#define NNODES 4096
#define KNB    16
#define K1     17
#define K1P    32
#define TT     16
#define NTT    16
#define DF     128
#define N_OUTER 3
#define N_INNER 5
#define EPS_RELF 0.1f
#define TINYF   1e-16f

typedef __attribute__((ext_vector_type(16))) __bf16 v16bf;
typedef __attribute__((ext_vector_type(8)))  float  v8f;

static __device__ __forceinline__ v8f wmma_bf16(v16bf a, v16bf b, v8f c) {
  // D(16x16 f32) = A(16x32 bf16) * B(32x16 bf16) + C
  return __builtin_amdgcn_wmma_f32_16x16x32_bf16(false, a, false, b, (short)0, c,
                                                 false, false);
}

// element index -> K index for 16-bit A (16x32) operand, contiguous order:
// lanes 0-15: elems 0-15 -> K0..15
// lanes16-31: elems 0-15 -> K16..31
static __device__ __forceinline__ int amap(int i, int h) {
  return i + 16 * h;
}

__global__ void __launch_bounds__(512)
ltfgw_kernel(const float* __restrict__ x,          // (N, D)
             const float* __restrict__ adj,        // (N, N)
             const int*   __restrict__ neighbors,  // (N, K)
             const float* __restrict__ tmpl,       // (T, NT, NT)
             const float* __restrict__ tfeat,      // (T, NT, D)
             const float* __restrict__ q0,         // (T, NT)
             const float* __restrict__ alpha0,     // (1,)
             float* __restrict__ out)              // (N, T)
{
  __shared__ float  C1f[K1P][K1P];   // padded adjacency sub-matrix (f32)
  __shared__ __bf16 C1b[K1P][K1P];   // same, bf16 for WMMA A operand
  __shared__ __bf16 Fib[K1P][DF];    // gathered features, bf16
  __shared__ __bf16 Gs[TT][K1P][NTT];// per-template transport plan (bf16)
  __shared__ __bf16 Hs[TT][K1P][NTT];// per-template C1*G intermediate (bf16)
  __shared__ float  fisq[K1P];       // row |Fi|^2
  __shared__ float  cCrow[K1P];      // sum_j C1[i][j]^2 * p[j]
  __shared__ int    nbS[K1P];

  const int n    = blockIdx.x;
  const int tid  = threadIdx.x;
  const int t    = tid >> 5;    // wave id == template id
  const int lane = tid & 31;
  const int h    = lane >> 4;   // half-wave
  const int nn   = lane & 15;   // column (N index) in all tiles

  // ---- neighborhood index list (self + K neighbors, padded) ----
  if (tid < K1P) {
    int v = 0;
    if (tid == 0)      v = n;
    else if (tid < K1) v = neighbors[n * KNB + (tid - 1)];
    nbS[tid] = v;
  }
  __syncthreads();

  // ---- gather Fi -> bf16 LDS + row squared norms (rows >= K1 zeroed) ----
  {
    const int row = 2 * t + h;           // each half-wave owns one row
    const int src = nbS[row];
    float s = 0.f;
    #pragma unroll
    for (int j = 0; j < 8; ++j) {
      const int d = nn * 8 + j;
      const float xv = (row < K1) ? x[(long)src * DF + d] : 0.f;
      Fib[row][d] = (__bf16)xv;
      s += xv * xv;
    }
    #pragma unroll
    for (int m = 1; m < 16; m <<= 1) s += __shfl_xor(s, m, 32);
    if (nn == 0) fisq[row] = s;
  }
  // ---- gather C1 (zero padded) ----
  for (int e = tid; e < K1P * K1P; e += 512) {
    const int i = e >> 5, j = e & 31;
    float a = 0.f;
    if (i < K1 && j < K1) a = adj[(long)nbS[i] * NNODES + nbS[j]];
    C1f[i][j] = a;
    C1b[i][j] = (__bf16)a;
  }
  __syncthreads();

  // ---- cCrow[i] = sum_j C1[i][j]^2 * p[j],  p = 1/17 on first 17 entries ----
  {
    const int row = 2 * t + h;
    const float c0 = C1f[row][nn];
    const float c1 = C1f[row][nn + 16];
    float s = c0 * c0 * (1.0f / K1) + ((nn + 16) < K1 ? c1 * c1 * (1.0f / K1) : 0.f);
    #pragma unroll
    for (int m = 1; m < 16; m <<= 1) s += __shfl_xor(s, m, 32);
    if (nn == 0) cCrow[row] = s;
  }
  __syncthreads();

  // ---- per-wave template constants (no workspace needed) ----
  const float alpha = 1.0f / (1.0f + __expf(-alpha0[0]));

  // softmax over q0[t, :] : lane holds m = nn (both halves replicate)
  float qn;
  {
    const float qv = q0[t * NTT + nn];
    float mx = qv;
    #pragma unroll
    for (int m = 1; m < 16; m <<= 1) mx = fmaxf(mx, __shfl_xor(mx, m, 32));
    float e = __expf(qv - mx), sum = e;
    #pragma unroll
    for (int m = 1; m < 16; m <<= 1) sum += __shfl_xor(sum, m, 32);
    qn = e / sum;
  }

  // c2B: B operand = C2^T padded to 32x16  (b[i] = C2[t][nn][i] for K<16)
  // qc2n = sum_m q[t][m] * C2[t][nn][m]^2
  v16bf c2B;
  float qc2n = 0.f;
  {
    const float* crow = tmpl + ((long)(t * NTT + nn)) * NTT;  // C2[t][nn][:]
    #pragma unroll
    for (int i = 0; i < 16; ++i) {
      const float c = crow[i];
      const float qi = __shfl(qn, i, 32);  // q[t][i] lives in lane i
      qc2n += qi * c * c;
      c2B[i] = (h == 0) ? (__bf16)c : (__bf16)0.0f;
    }
  }

  // ---- cross term: Fi(32x128) @ tfeat[t]^T(128x16) via 8 WMMAs;
  //      also accumulate |tfeat[t][nn]|^2 along the way ----
  v8f acc0 = {}, acc1 = {};
  float tfsqn = 0.f;
  #pragma unroll
  for (int ch = 0; ch < 4; ++ch) {
    v16bf bop, a0, a1;
    const float* tb = tfeat + ((long)(t * NTT + nn)) * DF + ch * 32 + h * 16;
    #pragma unroll
    for (int i = 0; i < 16; ++i) {
      const float v = tb[i];
      tfsqn += v * v;
      bop[i] = (__bf16)v;
    }
    #pragma unroll
    for (int i = 0; i < 16; ++i) {
      const int k = ch * 32 + amap(i, h);
      a0[i] = Fib[nn][k];
      a1[i] = Fib[16 + nn][k];
    }
    acc0 = wmma_bf16(a0, bop, acc0);
    acc1 = wmma_bf16(a1, bop, acc1);
  }
  tfsqn += __shfl_xor(tfsqn, 16, 32);  // two halves cover disjoint d-ranges

  // ---- M in C/D register layout: row = r + 8h (tile0) / 16 + r + 8h (tile1) ----
  float M0[8], M1[8];
  #pragma unroll
  for (int r = 0; r < 8; ++r) {
    M0[r] = fisq[r + 8 * h]      + tfsqn - 2.0f * acc0[r];
    M1[r] = fisq[16 + r + 8 * h] + tfsqn - 2.0f * acc1[r];
  }

  // ---- G0 = p (x) q, padded rows zero ----
  #pragma unroll
  for (int r = 0; r < 8; ++r) {
    const int i0 = r + 8 * h, i1 = i0 + 16;
    Gs[t][i0][nn] = (__bf16)((1.0f / K1) * qn);
    Gs[t][i1][nn] = (__bf16)((i1 < K1 ? (1.0f / K1) : 0.0f) * qn);
  }

  float pk0[8], pk1[8];
  #pragma unroll
  for (int r = 0; r < 8; ++r) {
    const int i1 = r + 8 * h + 16;
    pk0[r] = 1.0f / K1;                        // rows 0..15 all real
    pk1[r] = (i1 < K1) ? (1.0f / K1) : 0.0f;   // only row 16 real
  }

  float g0[8], g1[8], Gf0[8], Gf1[8];

  // grad = (1-alpha)*M + alpha*(cC - 2*C1*G*C2^T)  -- 4 WMMAs
  auto compute_grad = [&]() {
    v16bf bop, a0, a1;
    #pragma unroll
    for (int i = 0; i < 16; ++i) bop[i] = Gs[t][h * 16 + i][nn];   // B = G (32x16)
    #pragma unroll
    for (int i = 0; i < 16; ++i) {
      const int k = amap(i, h);
      a0[i] = C1b[nn][k];
      a1[i] = C1b[16 + nn][k];
    }
    v8f h0 = {}, h1 = {};
    h0 = wmma_bf16(a0, bop, h0);       // H = C1 * G (32x16)
    h1 = wmma_bf16(a1, bop, h1);
    #pragma unroll
    for (int r = 0; r < 8; ++r) {
      Hs[t][r + 8 * h][nn]      = (__bf16)h0[r];
      Hs[t][16 + r + 8 * h][nn] = (__bf16)h1[r];
    }
    v16bf ha0, ha1;                    // A = H with K padded 16->32 by zeros
    #pragma unroll
    for (int i = 0; i < 16; ++i) {
      const int k = amap(i, h);
      const __bf16 z = (__bf16)0.0f;
      ha0[i] = (k < 16) ? Hs[t][nn][k]      : z;
      ha1[i] = (k < 16) ? Hs[t][16 + nn][k] : z;
    }
    v8f r0 = {}, r1 = {};
    r0 = wmma_bf16(ha0, c2B, r0);      // (C1*G) * C2^T
    r1 = wmma_bf16(ha1, c2B, r1);
    #pragma unroll
    for (int r = 0; r < 8; ++r) {
      const int i0 = r + 8 * h, i1 = i0 + 16;
      g0[r] = (1.0f - alpha) * M0[r] + alpha * (cCrow[i0] + qc2n - 2.0f * r0[r]);
      g1[r] = (1.0f - alpha) * M1[r] + alpha * (cCrow[i1] + qc2n - 2.0f * r1[r]);
    }
  };

  // Sinkhorn (registers + wave32 shuffles) and new G
  auto sink_update = [&]() {
    // lo/hi over the REAL 17x16 region only
    float lo = 3.4e38f, hi = -3.4e38f;
    #pragma unroll
    for (int r = 0; r < 8; ++r) { lo = fminf(lo, g0[r]); hi = fmaxf(hi, g0[r]); }
    if (h == 0) { lo = fminf(lo, g1[0]); hi = fmaxf(hi, g1[0]); }  // row 16
    #pragma unroll
    for (int m = 1; m < 32; m <<= 1) {
      lo = fminf(lo, __shfl_xor(lo, m, 32));
      hi = fmaxf(hi, __shfl_xor(hi, m, 32));
    }
    const float inveps = 1.0f / (EPS_RELF * (hi - lo) + TINYF);
    float Kr0[8], Kr1[8];
    #pragma unroll
    for (int r = 0; r < 8; ++r) {
      Kr0[r] = __expf(-(g0[r] - lo) * inveps);
      Kr1[r] = __expf(-(g1[r] - lo) * inveps);
    }
    float u0[8], u1[8], v = 1.0f;
    for (int it = 0; it < N_INNER; ++it) {
      // u = p / max(K v): row-sums over 16 columns (half-wave reductions)
      #pragma unroll
      for (int r = 0; r < 8; ++r) {
        float s0 = Kr0[r] * v, s1 = Kr1[r] * v;
        #pragma unroll
        for (int m = 1; m < 16; m <<= 1) {
          s0 += __shfl_xor(s0, m, 32);
          s1 += __shfl_xor(s1, m, 32);
        }
        u0[r] = pk0[r] / fmaxf(s0, TINYF);
        u1[r] = pk1[r] / fmaxf(s1, TINYF);
      }
      // v = q / max(K^T u): column-sums over 32 rows (halves combine via xor16)
      float c = 0.f;
      #pragma unroll
      for (int r = 0; r < 8; ++r) c += Kr0[r] * u0[r] + Kr1[r] * u1[r];
      c += __shfl_xor(c, 16, 32);
      v = qn / fmaxf(c, TINYF);
    }
    #pragma unroll
    for (int r = 0; r < 8; ++r) {
      Gf0[r] = u0[r] * Kr0[r] * v;
      Gf1[r] = u1[r] * Kr1[r] * v;
      Gs[t][r + 8 * h][nn]      = (__bf16)Gf0[r];
      Gs[t][16 + r + 8 * h][nn] = (__bf16)Gf1[r];
    }
  };

  for (int o = 0; o < N_OUTER; ++o) {
    compute_grad();
    sink_update();
  }
  compute_grad();  // final gw with final G

  float s = 0.f;
  #pragma unroll
  for (int r = 0; r < 8; ++r) s += g0[r] * Gf0[r] + g1[r] * Gf1[r];
  #pragma unroll
  for (int m = 1; m < 32; m <<= 1) s += __shfl_xor(s, m, 32);
  if (lane == 0) out[n * TT + t] = s;
}

extern "C" void kernel_launch(void* const* d_in, const int* in_sizes, int n_in,
                              void* d_out, int out_size, void* d_ws, size_t ws_size,
                              hipStream_t stream) {
  (void)in_sizes; (void)n_in; (void)out_size; (void)d_ws; (void)ws_size;
  const float* x    = (const float*)d_in[0];
  const float* adj  = (const float*)d_in[1];
  const int*   nbr  = (const int*)d_in[2];
  const float* tmpl = (const float*)d_in[3];
  const float* tf   = (const float*)d_in[4];
  const float* q0   = (const float*)d_in[5];
  const float* a0   = (const float*)d_in[6];
  float* out = (float*)d_out;

  ltfgw_kernel<<<NNODES, 512, 0, stream>>>(x, adj, nbr, tmpl, tf, q0, a0, out);
}
